// EfficientAttention_89678917141243
// MI455X (gfx1250) — matrix operand order repaired and checked by reading, not compiled, not run on hardware
//
#include <hip/hip_runtime.h>
#include <stdint.h>

// ---------------------------------------------------------------------------
// Performer-style linear attention for MI455X (gfx1250, wave32, WMMA).
// Projections: LDS-staged (async-to-LDS, double buffered) bf16 WMMA GEMM.
// ---------------------------------------------------------------------------

typedef __bf16 bf16_t;
typedef __attribute__((ext_vector_type(16))) __bf16 bf16x16;
typedef __attribute__((ext_vector_type(8)))  __bf16 bf16x8;
typedef __attribute__((ext_vector_type(8)))  float  f32x8;

constexpr int B_  = 4;
constexpr int S_  = 8192;
constexpr int D_  = 1024;
constexpr int H_  = 16;
constexpr int F_  = 64;
constexpr int DK_ = 64;
constexpr int BS_ = B_ * S_;
constexpr int BH_ = B_ * H_;
constexpr float EPS_ = 1e-6f;

#define DEVFN __device__ __forceinline__

// A fragment (16x32 bf16, MxK): lane holds row M=lane&15.
// half=0 lanes: elems 0..15 = K[k0..k0+15]
// half=1 lanes: elems 0..15 = K[k0+16..k0+31]
// caller passes p = rowPtr + k0 + half*16.
DEVFN bf16x16 ld_afrag(const bf16_t* p) {
  bf16x8 lo = *(const bf16x8*)(p);
  bf16x8 hi = *(const bf16x8*)(p + 8);
  bf16x16 r;
#pragma unroll
  for (int i = 0; i < 8; ++i) { r[i] = lo[i]; r[i + 8] = hi[i]; }
  return r;
}

// B fragment (32x16 bf16, KxN): lane holds column N=lane&15,
// 16 contiguous K values starting at k0 + half*16 (Bt is N x K row-major).
DEVFN bf16x16 ld_bfrag(const bf16_t* p) { return *(const bf16x16*)p; }

DEVFN f32x8 wmma_bf16(bf16x16 a, bf16x16 b, f32x8 c) {
  return __builtin_amdgcn_wmma_f32_16x16x32_bf16(
      /*neg_a=*/false, a, /*neg_b=*/false, b,
      /*c_mod=*/(short)0, c, /*reuse_a=*/false, /*reuse_b=*/false);
}

// Async copy 16 bytes global -> LDS (ASYNCcnt-tracked DMA path).
DEVFN void async_b128_to_lds(unsigned lds_byte_addr, const bf16_t* gsrc) {
  asm volatile("global_load_async_to_lds_b128 %0, %1, off"
               :: "v"(lds_byte_addr), "v"(gsrc) : "memory");
}
DEVFN void wait_asynccnt0() {
  asm volatile("s_wait_asynccnt 0x0" ::: "memory");
}

enum { OUT_F32 = 0, OUT_BF16 = 1, OUT_BF16_VT = 2 };

// ---------------------------------------------------------------------------
// Projection GEMM: C(MxN) = A(MxK bf16) * Bt(NxK bf16)^T + bias.
// 256 threads = 8 waves; block tile 256x64; wave tile 32x64 (8 WMMA / k-step).
// B tile (64x32 bf16 = 4KB) staged in double-buffered LDS via
// global_load_async_to_lds_b128; A fragments register-double-buffered.
// ---------------------------------------------------------------------------
template <int MODE>
__global__ __launch_bounds__(256) void gemm_async_kernel(
    const bf16_t* __restrict__ A, const bf16_t* __restrict__ Bt,
    const float* __restrict__ bias, void* __restrict__ Cout,
    int M, int N, int K) {
  __shared__ __align__(64) bf16_t sB[2][64 * 32];  // [buf][n][k] 2 x 4KB

  const int tid  = threadIdx.x;
  const int wave = tid >> 5;
  const int lane = tid & 31;
  const int rowA = lane & 15;
  const int hf   = lane >> 4;
  const int colB = lane & 15;

  const int m0 = blockIdx.x * 256 + wave * 32;  // wave rows [m0, m0+32)
  const int n0 = blockIdx.y * 64;

  // Per-thread async-copy slot: thread t moves 16B; row n = t>>2, k-chunk t&3.
  const int cn = tid >> 2;
  const int ck = (tid & 3) * 8;
  const bf16_t* gsrc = Bt + (size_t)(n0 + cn) * K + ck;
  const unsigned ldsDst0 = (unsigned)(uintptr_t)(&sB[0][0]) + (unsigned)tid * 16u;
  const unsigned bufBytes = 64 * 32 * 2;

  const bf16_t* arow0 = A + (size_t)(m0 + rowA) * K + hf * 16;
  const bf16_t* arow1 = A + (size_t)(m0 + 16 + rowA) * K + hf * 16;

  f32x8 c[2][4] = {};

  // Prologue: stage k-step 0, preload A fragments for step 0.
  async_b128_to_lds(ldsDst0, gsrc);
  bf16x16 a0 = ld_afrag(arow0);
  bf16x16 a1 = ld_afrag(arow1);

  const int nsteps = K / 32;
  for (int i = 0; i < nsteps; ++i) {
    wait_asynccnt0();   // my copy of step i is in LDS
    __syncthreads();    // everyone's copy of step i is in LDS; step i-1 reads done
    if (i + 1 < nsteps) {
      async_b128_to_lds(ldsDst0 + (unsigned)((i + 1) & 1) * bufBytes,
                        gsrc + (size_t)(i + 1) * 32);
    }
    // Register double-buffer for A: fetch step i+1 while computing step i.
    bf16x16 na0 = a0, na1 = a1;
    if (i + 1 < nsteps) {
      na0 = ld_afrag(arow0 + (i + 1) * 32);
      na1 = ld_afrag(arow1 + (i + 1) * 32);
    }
    const bf16_t* sb = &sB[i & 1][0];
#pragma unroll
    for (int t = 0; t < 4; ++t) {
      bf16x16 bf_ = *(const bf16x16*)(sb + (t * 16 + colB) * 32 + hf * 16);
      c[0][t] = wmma_bf16(a0, bf_, c[0][t]);
      c[1][t] = wmma_bf16(a1, bf_, c[1][t]);
    }
    a0 = na0; a1 = na1;
  }

#pragma unroll
  for (int g = 0; g < 2; ++g) {
    const int mg = m0 + g * 16;
#pragma unroll
    for (int t = 0; t < 4; ++t) {
      const int n = n0 + t * 16 + colB;
      const float bv = bias ? bias[n] : 0.0f;
#pragma unroll
      for (int r = 0; r < 8; ++r) {
        const int m = mg + hf * 8 + r;
        const float v = c[g][t][r] + bv;
        if (MODE == OUT_F32) {
          ((float*)Cout)[(size_t)m * N + n] = v;
        } else if (MODE == OUT_BF16) {
          ((bf16_t*)Cout)[(size_t)m * N + n] = (bf16_t)v;
        } else {  // OUT_BF16_VT: scatter into vT[b,h,dk,s]
          const int b = m >> 13, s = m & (S_ - 1);
          const int h = n >> 6,  dk = n & 63;
          ((bf16_t*)Cout)[(((size_t)b * H_ + h) * DK_ + dk) * S_ + s] = (bf16_t)v;
        }
      }
    }
  }
}

// ---------------------------------------------------------------------------
// phi: per head, (S x DK) * rwT(F x DK)^T -> relu + eps.
// TRANS=false -> qp[bh,s,f]; TRANS=true -> kpT[bh,f,s]. grid=(B*H, S/64).
// ---------------------------------------------------------------------------
template <bool TRANS>
__global__ __launch_bounds__(128) void phi_kernel(
    const bf16_t* __restrict__ qk, const bf16_t* __restrict__ rwT,
    bf16_t* __restrict__ outp) {
  const int bh = blockIdx.x;
  const int b = bh >> 4, h = bh & (H_ - 1);
  const int wave = threadIdx.x >> 5, lane = threadIdx.x & 31;
  const int rowA = lane & 15, hf = lane >> 4, colB = lane & 15;
  const int s0 = blockIdx.y * 64 + wave * 16;

  const bf16_t* arow = qk + ((size_t)b * S_ + s0 + rowA) * D_ + h * DK_ + hf * 16;
  const bf16_t* Bb   = rwT + (size_t)h * F_ * DK_;

  f32x8 c[4] = {};
#pragma unroll
  for (int k0 = 0; k0 < DK_; k0 += 32) {
    bf16x16 af = ld_afrag(arow + k0);
#pragma unroll
    for (int t = 0; t < 4; ++t) {
      const bf16_t* bp = Bb + (size_t)(t * 16 + colB) * DK_ + hf * 16 + k0;
      c[t] = wmma_bf16(af, ld_bfrag(bp), c[t]);
    }
  }

#pragma unroll
  for (int t = 0; t < 4; ++t) {
    const int f = t * 16 + colB;
#pragma unroll
    for (int r = 0; r < 8; ++r) {
      const int s = s0 + hf * 8 + r;
      const float v = fmaxf(c[t][r], 0.0f) + EPS_;
      if (!TRANS) outp[((size_t)bh * S_ + s) * F_ + f] = (bf16_t)v;
      else        outp[((size_t)bh * F_ + f) * S_ + s] = (bf16_t)v;
    }
  }
}

// ---------------------------------------------------------------------------
// kv = kp^T @ v (reduction over S), accumulated transposed into kvT[bh,dk,f].
// grid = (B*H, 16 chunks of S); f32 atomics into zeroed buffer.
// ---------------------------------------------------------------------------
__global__ __launch_bounds__(128) void kv_kernel(
    const bf16_t* __restrict__ kpT, const bf16_t* __restrict__ vT,
    float* __restrict__ kv32) {
  const int bh = blockIdx.x;
  const int wave = threadIdx.x >> 5, lane = threadIdx.x & 31;
  const int rowA = lane & 15, hf = lane >> 4, colB = lane & 15;
  const int f0 = wave * 16;
  const int kbase = blockIdx.y * (S_ / 16);

  const bf16_t* arow = kpT + ((size_t)bh * F_ + f0 + rowA) * S_ + hf * 16;
  const bf16_t* Bb   = vT + (size_t)bh * DK_ * S_;

  f32x8 c[4] = {};
  for (int k0 = kbase; k0 < kbase + S_ / 16; k0 += 32) {
    bf16x16 af = ld_afrag(arow + k0);
#pragma unroll
    for (int t = 0; t < 4; ++t) {
      const bf16_t* bp = Bb + (size_t)(t * 16 + colB) * S_ + hf * 16 + k0;
      c[t] = wmma_bf16(af, ld_bfrag(bp), c[t]);
    }
  }

#pragma unroll
  for (int t = 0; t < 4; ++t) {
    const int dk = t * 16 + colB;
#pragma unroll
    for (int r = 0; r < 8; ++r) {
      const int f = f0 + hf * 8 + r;
      atomicAdd(&kv32[((size_t)bh * DK_ + dk) * F_ + f], c[t][r]);
    }
  }
}

// ksum[bh,f] = sum_s kpT[bh,f,s]
__global__ __launch_bounds__(64) void ksum_kernel(
    const bf16_t* __restrict__ kpT, float* __restrict__ ksum32) {
  const int bh = blockIdx.x, f = threadIdx.x;
  const bf16_t* p = kpT + ((size_t)bh * F_ + f) * S_;
  float acc = 0.0f;
  for (int s = 0; s < S_; s += 8) {
    bf16x8 v = *(const bf16x8*)(p + s);
#pragma unroll
    for (int i = 0; i < 8; ++i) acc += (float)v[i];
  }
  ksum32[(size_t)bh * F_ + f] = acc;
}

// ---------------------------------------------------------------------------
// qkv = qp @ kv, norm = qp @ ksum (extra WMMA, ksum in B column 0),
// out = qkv / (norm + eps) stored bf16 in (B,S,D) layout. grid=(B*H, S/64).
// ---------------------------------------------------------------------------
__global__ __launch_bounds__(128) void qkv_kernel(
    const bf16_t* __restrict__ qp, const bf16_t* __restrict__ kvT,
    const bf16_t* __restrict__ ksum, bf16_t* __restrict__ outp) {
  const int bh = blockIdx.x;
  const int b = bh >> 4, h = bh & (H_ - 1);
  const int wave = threadIdx.x >> 5, lane = threadIdx.x & 31;
  const int rowA = lane & 15, hf = lane >> 4, colB = lane & 15;
  const int s0 = blockIdx.y * 64 + wave * 16;

  const bf16_t* arow = qp + ((size_t)bh * S_ + s0 + rowA) * F_ + hf * 16;
  const bf16_t* Bb   = kvT + (size_t)bh * DK_ * F_;
  const bf16_t* kss  = ksum + (size_t)bh * F_;

  f32x8 c[4] = {};
  f32x8 c2 = {};
#pragma unroll
  for (int k0 = 0; k0 < F_; k0 += 32) {
    bf16x16 af = ld_afrag(arow + k0);
#pragma unroll
    for (int t = 0; t < 4; ++t) {
      const bf16_t* bp = Bb + (size_t)(t * 16 + colB) * F_ + hf * 16 + k0;
      c[t] = wmma_bf16(af, ld_bfrag(bp), c[t]);
    }
    bf16x16 b2;
#pragma unroll
    for (int e = 0; e < 16; ++e) b2[e] = (bf16_t)0.0f;
    if (colB == 0) {
      const bf16_t* kp_ = kss + k0 + hf * 16;
#pragma unroll
      for (int e = 0; e < 16; ++e) b2[e] = kp_[e];
    }
    c2 = wmma_bf16(af, b2, c2);
  }

#pragma unroll
  for (int r = 0; r < 8; ++r) {
    const float nv = __shfl(c2[r], lane & 16, 32);  // norm for this row
    const int s = s0 + hf * 8 + r;
#pragma unroll
    for (int t = 0; t < 4; ++t) {
      const int dk = t * 16 + colB;
      const float v = c[t][r] / (nv + EPS_);
      outp[((size_t)b * S_ + s) * D_ + h * DK_ + dk] = (bf16_t)v;
    }
  }
}

// ----------------------------- small helpers -------------------------------
__global__ void f32_to_bf16_kernel(const float* __restrict__ in,
                                   bf16_t* __restrict__ outp, size_t n) {
  size_t i = (size_t)blockIdx.x * blockDim.x + threadIdx.x;
  const size_t stride = (size_t)gridDim.x * blockDim.x;
  for (; i < n; i += stride) outp[i] = (bf16_t)in[i];
}

// W (K=D x N=D) row-major -> Wt (N x D) bf16
__global__ void transpose_w_kernel(const float* __restrict__ w,
                                   bf16_t* __restrict__ wt) {
  const size_t i = (size_t)blockIdx.x * blockDim.x + threadIdx.x;
  if (i >= (size_t)D_ * D_) return;
  const int nn = (int)(i / D_), kk = (int)(i % D_);
  wt[i] = (bf16_t)w[(size_t)kk * D_ + nn];
}

// rw (H,DK,F) -> rwT (H,F,DK) bf16
__global__ void transpose_rw_kernel(const float* __restrict__ rw,
                                    bf16_t* __restrict__ rwT) {
  const size_t i = (size_t)blockIdx.x * blockDim.x + threadIdx.x;
  if (i >= (size_t)H_ * F_ * DK_) return;
  const int dk = (int)(i % DK_);
  const int f  = (int)((i / DK_) % F_);
  const int h  = (int)(i / ((size_t)F_ * DK_));
  rwT[i] = (bf16_t)rw[((size_t)h * DK_ + dk) * F_ + f];
}

// ---------------------------------------------------------------------------
extern "C" void kernel_launch(void* const* d_in, const int* in_sizes, int n_in,
                              void* d_out, int out_size, void* d_ws,
                              size_t ws_size, hipStream_t stream) {
  (void)in_sizes; (void)n_in; (void)out_size; (void)ws_size;
  const float* x  = (const float*)d_in[0];
  const float* Wq = (const float*)d_in[1];
  const float* bq = (const float*)d_in[2];
  const float* Wk = (const float*)d_in[3];
  const float* bk = (const float*)d_in[4];
  const float* Wv = (const float*)d_in[5];
  const float* bv = (const float*)d_in[6];
  const float* Wo = (const float*)d_in[7];
  const float* bo = (const float*)d_in[8];
  const float* rw = (const float*)d_in[9];
  float* y = (float*)d_out;

  char* ws = (char*)d_ws;
  size_t off = 0;
  auto take = [&](size_t bytes) -> char* {
    char* p = ws + off;
    off += (bytes + 255) & ~(size_t)255;
    return p;
  };

  bf16_t* xb   = (bf16_t*)take((size_t)BS_ * D_ * 2);  // also reused as "out"
  bf16_t* wqT  = (bf16_t*)take((size_t)D_ * D_ * 2);
  bf16_t* wkT  = (bf16_t*)take((size_t)D_ * D_ * 2);
  bf16_t* wvT  = (bf16_t*)take((size_t)D_ * D_ * 2);
  bf16_t* woT  = (bf16_t*)take((size_t)D_ * D_ * 2);
  bf16_t* rwT  = (bf16_t*)take((size_t)H_ * F_ * DK_ * 2);
  bf16_t* qb   = (bf16_t*)take((size_t)BS_ * D_ * 2);
  bf16_t* kb   = (bf16_t*)take((size_t)BS_ * D_ * 2);
  bf16_t* vT   = (bf16_t*)take((size_t)BH_ * DK_ * S_ * 2);
  bf16_t* qp   = (bf16_t*)take((size_t)BH_ * S_ * F_ * 2);
  bf16_t* kpT  = (bf16_t*)take((size_t)BH_ * F_ * S_ * 2);
  float*  kv32 = (float*)take((size_t)BH_ * DK_ * F_ * 4);
  float*  ks32 = (float*)take((size_t)BH_ * F_ * 4);
  bf16_t* kvb  = (bf16_t*)take((size_t)BH_ * DK_ * F_ * 2);
  bf16_t* ksb  = (bf16_t*)take((size_t)BH_ * F_ * 2);
  bf16_t* outb = xb;  // x no longer needed once qkv stage runs

  const dim3 blk128(128), blk256(256);

  // Stage 0: precision conversion / layout transforms
  f32_to_bf16_kernel<<<dim3(4096), blk256, 0, stream>>>(x, xb, (size_t)BS_ * D_);
  transpose_w_kernel<<<dim3((D_ * D_ + 255) / 256), blk256, 0, stream>>>(Wq, wqT);
  transpose_w_kernel<<<dim3((D_ * D_ + 255) / 256), blk256, 0, stream>>>(Wk, wkT);
  transpose_w_kernel<<<dim3((D_ * D_ + 255) / 256), blk256, 0, stream>>>(Wv, wvT);
  transpose_w_kernel<<<dim3((D_ * D_ + 255) / 256), blk256, 0, stream>>>(Wo, woT);
  transpose_rw_kernel<<<dim3((H_ * F_ * DK_ + 255) / 256), blk256, 0, stream>>>(rw, rwT);

  // Stage 1: Q/K/V projections (async-LDS WMMA GEMM)
  const dim3 gGrid(BS_ / 256, D_ / 64);
  gemm_async_kernel<OUT_BF16><<<gGrid, blk256, 0, stream>>>(xb, wqT, bq, qb, BS_, D_, D_);
  gemm_async_kernel<OUT_BF16><<<gGrid, blk256, 0, stream>>>(xb, wkT, bk, kb, BS_, D_, D_);
  gemm_async_kernel<OUT_BF16_VT><<<gGrid, blk256, 0, stream>>>(xb, wvT, bv, vT, BS_, D_, D_);

  // Stage 2: feature maps (WMMA)
  const dim3 pGrid(BH_, S_ / 64);
  phi_kernel<false><<<pGrid, blk128, 0, stream>>>(qb, rwT, qp);
  phi_kernel<true><<<pGrid, blk128, 0, stream>>>(kb, rwT, kpT);

  // Stage 3: kv reduction + ksum
  hipMemsetAsync(kv32, 0, (size_t)BH_ * DK_ * F_ * 4, stream);
  ksum_kernel<<<dim3(BH_), dim3(64), 0, stream>>>(kpT, ks32);
  kv_kernel<<<dim3(BH_, 16), blk128, 0, stream>>>(kpT, vT, kv32);
  f32_to_bf16_kernel<<<dim3(256), blk256, 0, stream>>>(kv32, kvb, (size_t)BH_ * DK_ * F_);
  f32_to_bf16_kernel<<<dim3(16), blk256, 0, stream>>>(ks32, ksb, (size_t)BH_ * F_);

  // Stage 4: qkv + normalization (WMMA)
  qkv_kernel<<<pGrid, blk128, 0, stream>>>(qp, kvb, ksb, outb);

  // Stage 5: output projection -> f32 d_out (WMMA)
  gemm_async_kernel<OUT_F32><<<gGrid, blk256, 0, stream>>>(outb, woT, bo, y, BS_, D_, D_);
}
